// CirculantLinear_48713519071957
// MI455X (gfx1250) — matrix operand order repaired and checked by reading, not compiled, not run on hardware
//
#include <hip/hip_runtime.h>
#include <hip/hip_bf16.h>
#include <stdint.h>

// ---------------------------------------------------------------------------
// CirculantLinear == dense GEMM  y[b,o] = sum_k x[b,k] * c[o,(IN-k)%IN] + bias
// MI455X: compute-bound (AI ~ 860 flop/byte). bf16x3 split-precision WMMA
// (v_wmma_f32_16x16x32_bf16) gives ~fp32 accuracy at bf16 matrix throughput.
// Round 2 -> 3: replace {global->VGPR->LDS, 2 barriers} staging with CDNA5
// GLOBAL_LOAD_ASYNC_TO_LDS_B128 + 3-stage LDS pipeline (ASYNCcnt-tracked),
// one barrier per k-step, copies for tiles i+1/i+2 hidden under WMMA of tile i.
// ---------------------------------------------------------------------------

typedef __attribute__((ext_vector_type(16))) __bf16        v16bf;
typedef __attribute__((ext_vector_type(8)))  float         v8f;
typedef __attribute__((ext_vector_type(4)))  unsigned int  u32x4;

#define BM    128   // batch rows per block
#define BN    128   // output cols per block
#define BK    32    // k-slice per iteration (== WMMA K)
#define KPAD  40    // LDS row pitch in ushorts (80B): lane*20 mod 64 banks is
                    // a permutation -> conflict-free strided b128 fragment reads
#define PLANE   (BM * KPAD)           // ushorts per plane (Xh/Xl/Rh/Rl)
#define PLANE_B (PLANE * 2)           // bytes per plane
#define STAGE_B (4 * PLANE_B)         // bytes per pipeline stage (40 KB)
#define NSTAGE  3                     // pipeline depth (120 KB LDS total)

// Split fp32 into bf16 hi (RNE) + bf16 lo (truncated residual).
__device__ __forceinline__ void split_bf16(float v, unsigned short& h, unsigned short& l) {
  unsigned ub = __float_as_uint(v);
  unsigned r  = ub + 0x7FFFu + ((ub >> 16) & 1u);      // round-to-nearest-even
  unsigned hb = r & 0xFFFF0000u;
  float    lf = v - __uint_as_float(hb);               // exact residual
  h = (unsigned short)(hb >> 16);
  l = (unsigned short)(__float_as_uint(lf) >> 16);     // truncate residual to bf16
}

union FragCvt { u32x4 u[2]; v16bf v; };

// 32B global -> LDS async copy (2 x b128). The instruction offset advances
// both the LDS and global addresses (ISA 08 4.4), so one VGPR pair serves both
// halves. GVS mode: saddr = plane base (SGPR64), vaddr = per-lane byte offset.
__device__ __forceinline__ void async_copy32(unsigned lds, unsigned goff,
                                             const unsigned short* sbase) {
  asm volatile(
      "global_load_async_to_lds_b128 %0, %1, %2 offset:0\n\t"
      "global_load_async_to_lds_b128 %0, %1, %2 offset:16"
      :: "v"(lds), "v"(goff), "s"(sbase) : "memory");
}

// ---------------------------------------------------------------------------
// Pre-pass 1: x (fp32) -> xh/xl bf16 planes, 8 elements per thread.
// ---------------------------------------------------------------------------
extern "C" __global__ __launch_bounds__(256)
void split_x_kernel(const float* __restrict__ x,
                    unsigned short* __restrict__ xh,
                    unsigned short* __restrict__ xl)
{
  const size_t i = ((size_t)blockIdx.x * 256 + threadIdx.x) * 8;
  float4 a = *(const float4*)(x + i);
  float4 b = *(const float4*)(x + i + 4);
  const float v[8] = { a.x, a.y, a.z, a.w, b.x, b.y, b.z, b.w };
  u32x4 H, L;
  #pragma unroll
  for (int j = 0; j < 4; ++j) {
    unsigned short h0, l0, h1, l1;
    split_bf16(v[2*j],   h0, l0);
    split_bf16(v[2*j+1], h1, l1);
    H[j] = (unsigned)h0 | ((unsigned)h1 << 16);
    L[j] = (unsigned)l0 | ((unsigned)l1 << 16);
  }
  *(u32x4*)(xh + i) = H;
  *(u32x4*)(xl + i) = L;
}

// ---------------------------------------------------------------------------
// Pre-pass 2: R[o,k] = c[o, (IN-k)%IN] -> rh/rl bf16 planes (remap baked in).
// grid = (IN/2048, OUT), 8 elements per thread.
// ---------------------------------------------------------------------------
extern "C" __global__ __launch_bounds__(256)
void split_r_kernel(const float* __restrict__ c,
                    unsigned short* __restrict__ rh,
                    unsigned short* __restrict__ rl,
                    int IN)
{
  const int    o     = blockIdx.y;
  const int    kbase = (blockIdx.x * 256 + threadIdx.x) * 8;
  const float* crow  = c + (size_t)o * IN;
  float v[8];
  #pragma unroll
  for (int j = 0; j < 8; ++j) {
    const int k   = kbase + j;
    const int src = k ? (IN - k) : 0;       // circulant row of C_o
    v[j] = crow[src];
  }
  u32x4 H, L;
  #pragma unroll
  for (int j = 0; j < 4; ++j) {
    unsigned short h0, l0, h1, l1;
    split_bf16(v[2*j],   h0, l0);
    split_bf16(v[2*j+1], h1, l1);
    H[j] = (unsigned)h0 | ((unsigned)h1 << 16);
    L[j] = (unsigned)l0 | ((unsigned)l1 << 16);
  }
  const size_t dst = (size_t)o * IN + kbase;
  *(u32x4*)(rh + dst) = H;
  *(u32x4*)(rl + dst) = L;
}

// ---------------------------------------------------------------------------
// Main GEMM: async global->LDS pipeline (depth 3) + bf16x3 WMMA.
// ---------------------------------------------------------------------------
extern "C" __global__ __launch_bounds__(256, 1)
void circulant_wmma_async(const unsigned short* __restrict__ xh,
                          const unsigned short* __restrict__ xl,
                          const unsigned short* __restrict__ rh,
                          const unsigned short* __restrict__ rl,
                          const float* __restrict__ bias,
                          float* __restrict__ out,
                          int OUT, int IN)
{
  __shared__ unsigned short S[NSTAGE * 4 * PLANE];

  const int t      = threadIdx.x;
  const int wave   = t >> 5;
  const int lane   = t & 31;
  const int wm     = wave & 3;         // 4 wave-rows  -> 32-row slab each
  const int wn     = wave >> 2;        // 2 wave-cols  -> 64-col slab each
  const int laneLo = lane & 15;
  const int laneHi = lane >> 4;

  const int m0 = blockIdx.y * BM;
  const int n0 = blockIdx.x * BN;

  const int lrow = t >> 1;             // 0..127: row this thread copies
  const int lcc  = (t & 1) << 4;       // 0 or 16 ushorts within the k-slice

  // per-thread byte offsets (k-independent parts)
  const unsigned gx   = (unsigned)(((size_t)(m0 + lrow) * IN + lcc) * 2);
  const unsigned gr   = (unsigned)(((size_t)(n0 + lrow) * IN + lcc) * 2);
  const unsigned tlds = (unsigned)((lrow * KPAD + lcc) * 2);
  // low 32 bits of a generic LDS pointer == LDS byte offset (flat aperture)
  const unsigned ldsBase = (unsigned)(uintptr_t)(&S[0]);

  const v8f vzero = {};
  v8f acc[2][4];
  #pragma unroll
  for (int i = 0; i < 2; ++i)
    #pragma unroll
    for (int j = 0; j < 4; ++j) acc[i][j] = vzero;

  const int ntiles = IN / BK;

  // issue one tile (8 async b128 per thread -> ASYNCcnt += 8 per wave)
  auto issue_tile = [&](int ktile, int stage) {
    const unsigned koff  = (unsigned)(ktile * (BK * 2));
    const unsigned sbase = ldsBase + (unsigned)stage * STAGE_B + tlds;
    async_copy32(sbase + 0u * PLANE_B, gx + koff, xh);
    async_copy32(sbase + 1u * PLANE_B, gx + koff, xl);
    async_copy32(sbase + 2u * PLANE_B, gr + koff, rh);
    async_copy32(sbase + 3u * PLANE_B, gr + koff, rl);
  };

  // pipeline prologue: tiles 0 and 1 in flight
  issue_tile(0, 0);
  issue_tile(1, 1);

  int cs = 0;   // stage holding tile i
  for (int i = 0; i < ntiles; ++i) {
    // my wave's tile-i copies done (16 outstanding -> drain to 8 oldest-first)
    asm volatile("s_wait_asynccnt 8" ::: "memory");
    // all waves' tile-i data landed; stage being refilled is no longer read
    __syncthreads();

    // keep exactly two tiles in flight: issue tile i+2 (wrapped -> junk tile
    // into a stage that is never consumed; keeps ASYNCcnt bookkeeping uniform)
    int kpre = i + 2;
    if (kpre >= ntiles) kpre = 0;
    int is = cs + 2; if (is >= NSTAGE) is -= NSTAGE;
    issue_tile(kpre, is);

    // ---- LDS -> WMMA fragments (ISA 7.12.2 layouts) ----
    const unsigned short* st  = &S[cs * 4 * PLANE];
    const unsigned short* sXh = st;
    const unsigned short* sXl = st + PLANE;
    const unsigned short* sRh = st + 2 * PLANE;
    const unsigned short* sRl = st + 3 * PLANE;

    // A 16x32: lane L row M=L%16; lanes 0-15 K{0..7,16..23}, 16-31 K{8..15,24..31}
    v16bf ah[2], al[2];
    #pragma unroll
    for (int m16 = 0; m16 < 2; ++m16) {
      const unsigned short* pa = sXh + (wm*32 + m16*16 + laneLo) * KPAD;
      const unsigned short* qa = sXl + (wm*32 + m16*16 + laneLo) * KPAD;
      const int c0 = laneHi * 8;
      const int c1 = 16 + laneHi * 8;
      FragCvt f, g;
      f.u[0] = *(const u32x4*)(pa + c0);  f.u[1] = *(const u32x4*)(pa + c1);
      g.u[0] = *(const u32x4*)(qa + c0);  g.u[1] = *(const u32x4*)(qa + c1);
      ah[m16] = f.v;  al[m16] = g.v;
    }
    // B 32x16: lane n col N=n%16; lanes 0-15 K{0..7,16..23}, 16-31 K{8..15,24..31}
    v16bf bh[4], bl[4];
    #pragma unroll
    for (int n16 = 0; n16 < 4; ++n16) {
      const unsigned short* pb = sRh + (wn*64 + n16*16 + laneLo) * KPAD;
      const unsigned short* qb = sRl + (wn*64 + n16*16 + laneLo) * KPAD;
      const int c0 = laneHi * 8;
      const int c1 = 16 + laneHi * 8;
      FragCvt f, g;
      f.u[0] = *(const u32x4*)(pb + c0);  f.u[1] = *(const u32x4*)(pb + c1);
      g.u[0] = *(const u32x4*)(qb + c0);  g.u[1] = *(const u32x4*)(qb + c1);
      bh[n16] = f.v;  bl[n16] = g.v;
    }

    // ---- bf16x3: hh + hl + lh ----
    #pragma unroll
    for (int m16 = 0; m16 < 2; ++m16) {
      #pragma unroll
      for (int n16 = 0; n16 < 4; ++n16) {
        acc[m16][n16] = __builtin_amdgcn_wmma_f32_16x16x32_bf16(
            false, ah[m16], false, bh[n16], (short)0, acc[m16][n16], false, false);
        acc[m16][n16] = __builtin_amdgcn_wmma_f32_16x16x32_bf16(
            false, ah[m16], false, bl[n16], (short)0, acc[m16][n16], false, false);
        acc[m16][n16] = __builtin_amdgcn_wmma_f32_16x16x32_bf16(
            false, al[m16], false, bh[n16], (short)0, acc[m16][n16], false, false);
      }
    }

    cs = (cs + 1 == NSTAGE) ? 0 : cs + 1;
  }

  // ---- epilogue: bias + store (C/D layout: lane v -> M=v+8*laneHi, N=lane&15)
  #pragma unroll
  for (int n16 = 0; n16 < 4; ++n16) {
    const int   col = n0 + wn*64 + n16*16 + laneLo;
    const float bv  = bias[col];
    #pragma unroll
    for (int m16 = 0; m16 < 2; ++m16) {
      const int rbase = m0 + wm*32 + m16*16 + laneHi*8;
      #pragma unroll
      for (int v = 0; v < 8; ++v) {
        out[(size_t)(rbase + v) * OUT + col] = acc[m16][n16][v] + bv;
      }
    }
  }
}

// ---------------------------------------------------------------------------
// Fallback (workspace too small): fused in-loop conversion.
// ---------------------------------------------------------------------------
extern "C" __global__ __launch_bounds__(256, 1)
void circulant_wmma_fused(const float* __restrict__ x,
                          const float* __restrict__ c,
                          const float* __restrict__ bias,
                          float* __restrict__ out,
                          int OUT, int IN)
{
  __shared__ unsigned short Xh[BM * KPAD], Xl[BM * KPAD];
  __shared__ unsigned short Rh[BN * KPAD], Rl[BN * KPAD];

  const int t      = threadIdx.x;
  const int wave   = t >> 5;
  const int lane   = t & 31;
  const int wm     = wave & 3;
  const int wn     = wave >> 2;
  const int laneLo = lane & 15;
  const int laneHi = lane >> 4;

  const int m0 = blockIdx.y * BM;
  const int n0 = blockIdx.x * BN;

  const int lrow = t >> 1;
  const int lcc  = (t & 1) << 4;

  const float* __restrict__ xrow = x + (size_t)(m0 + lrow) * IN + lcc;
  const float* __restrict__ crow = c + (size_t)(n0 + lrow) * IN;
  unsigned short* XhP = &Xh[lrow * KPAD + lcc];
  unsigned short* XlP = &Xl[lrow * KPAD + lcc];
  unsigned short* RhP = &Rh[lrow * KPAD + lcc];
  unsigned short* RlP = &Rl[lrow * KPAD + lcc];

  const v8f vzero = {};
  v8f acc[2][4];
  #pragma unroll
  for (int i = 0; i < 2; ++i)
    #pragma unroll
    for (int j = 0; j < 4; ++j) acc[i][j] = vzero;

  for (int k0 = 0; k0 < IN; k0 += BK) {
    float4 a0 = *(const float4*)(xrow + k0 + 0);
    float4 a1 = *(const float4*)(xrow + k0 + 4);
    float4 a2 = *(const float4*)(xrow + k0 + 8);
    float4 a3 = *(const float4*)(xrow + k0 + 12);
    float xv[16] = { a0.x,a0.y,a0.z,a0.w, a1.x,a1.y,a1.z,a1.w,
                     a2.x,a2.y,a2.z,a2.w, a3.x,a3.y,a3.z,a3.w };
    float rv[16];
    #pragma unroll
    for (int j = 0; j < 16; ++j) {
      int k   = k0 + lcc + j;
      int src = k ? (IN - k) : 0;
      rv[j] = crow[src];
    }
    unsigned xhp[8], xlp[8], rhp[8], rlp[8];
    #pragma unroll
    for (int j = 0; j < 8; ++j) {
      unsigned short h0, l0, h1, l1;
      split_bf16(xv[2*j], h0, l0);  split_bf16(xv[2*j+1], h1, l1);
      xhp[j] = (unsigned)h0 | ((unsigned)h1 << 16);
      xlp[j] = (unsigned)l0 | ((unsigned)l1 << 16);
      split_bf16(rv[2*j], h0, l0);  split_bf16(rv[2*j+1], h1, l1);
      rhp[j] = (unsigned)h0 | ((unsigned)h1 << 16);
      rlp[j] = (unsigned)l0 | ((unsigned)l1 << 16);
    }
    __syncthreads();
    u32x4 s;
    s[0]=xhp[0]; s[1]=xhp[1]; s[2]=xhp[2]; s[3]=xhp[3]; ((u32x4*)XhP)[0] = s;
    s[0]=xhp[4]; s[1]=xhp[5]; s[2]=xhp[6]; s[3]=xhp[7]; ((u32x4*)XhP)[1] = s;
    s[0]=xlp[0]; s[1]=xlp[1]; s[2]=xlp[2]; s[3]=xlp[3]; ((u32x4*)XlP)[0] = s;
    s[0]=xlp[4]; s[1]=xlp[5]; s[2]=xlp[6]; s[3]=xlp[7]; ((u32x4*)XlP)[1] = s;
    s[0]=rhp[0]; s[1]=rhp[1]; s[2]=rhp[2]; s[3]=rhp[3]; ((u32x4*)RhP)[0] = s;
    s[0]=rhp[4]; s[1]=rhp[5]; s[2]=rhp[6]; s[3]=rhp[7]; ((u32x4*)RhP)[1] = s;
    s[0]=rlp[0]; s[1]=rlp[1]; s[2]=rlp[2]; s[3]=rlp[3]; ((u32x4*)RlP)[0] = s;
    s[0]=rlp[4]; s[1]=rlp[5]; s[2]=rlp[6]; s[3]=rlp[7]; ((u32x4*)RlP)[1] = s;
    __syncthreads();

    v16bf ah[2], al[2];
    #pragma unroll
    for (int m16 = 0; m16 < 2; ++m16) {
      const unsigned short* pa = &Xh[(wm*32 + m16*16 + laneLo) * KPAD];
      const unsigned short* qa = &Xl[(wm*32 + m16*16 + laneLo) * KPAD];
      const int c0 = laneHi * 8;
      const int c1 = 16 + laneHi * 8;
      FragCvt f, g;
      f.u[0] = *(const u32x4*)(pa + c0);  f.u[1] = *(const u32x4*)(pa + c1);
      g.u[0] = *(const u32x4*)(qa + c0);  g.u[1] = *(const u32x4*)(qa + c1);
      ah[m16] = f.v;  al[m16] = g.v;
    }
    v16bf bh[4], bl[4];
    #pragma unroll
    for (int n16 = 0; n16 < 4; ++n16) {
      const unsigned short* pb = &Rh[(wn*64 + n16*16 + laneLo) * KPAD];
      const unsigned short* qb = &Rl[(wn*64 + n16*16 + laneLo) * KPAD];
      const int c0 = laneHi * 8;
      const int c1 = 16 + laneHi * 8;
      FragCvt f, g;
      f.u[0] = *(const u32x4*)(pb + c0);  f.u[1] = *(const u32x4*)(pb + c1);
      g.u[0] = *(const u32x4*)(qb + c0);  g.u[1] = *(const u32x4*)(qb + c1);
      bh[n16] = f.v;  bl[n16] = g.v;
    }
    #pragma unroll
    for (int m16 = 0; m16 < 2; ++m16) {
      #pragma unroll
      for (int n16 = 0; n16 < 4; ++n16) {
        acc[m16][n16] = __builtin_amdgcn_wmma_f32_16x16x32_bf16(
            false, ah[m16], false, bh[n16], (short)0, acc[m16][n16], false, false);
        acc[m16][n16] = __builtin_amdgcn_wmma_f32_16x16x32_bf16(
            false, ah[m16], false, bl[n16], (short)0, acc[m16][n16], false, false);
        acc[m16][n16] = __builtin_amdgcn_wmma_f32_16x16x32_bf16(
            false, al[m16], false, bh[n16], (short)0, acc[m16][n16], false, false);
      }
    }
  }

  #pragma unroll
  for (int n16 = 0; n16 < 4; ++n16) {
    const int   col = n0 + wn*64 + n16*16 + laneLo;
    const float bv  = bias[col];
    #pragma unroll
    for (int m16 = 0; m16 < 2; ++m16) {
      const int rbase = m0 + wm*32 + m16*16 + laneHi*8;
      #pragma unroll
      for (int v = 0; v < 8; ++v) {
        out[(size_t)(rbase + v) * OUT + col] = acc[m16][n16][v] + bv;
      }
    }
  }
}

extern "C" void kernel_launch(void* const* d_in, const int* in_sizes, int n_in,
                              void* d_out, int out_size, void* d_ws, size_t ws_size,
                              hipStream_t stream) {
  const float* x    = (const float*)d_in[0];
  const float* c    = (const float*)d_in[1];
  const float* bias = (const float*)d_in[2];
  float*       out  = (float*)d_out;

  const int OUT = in_sizes[2];              // bias length
  const int IN  = in_sizes[1] / OUT;        // c is OUT x IN
  const int Bb  = in_sizes[0] / IN;         // x is B x IN

  const dim3 grid(OUT / BN, Bb / BM);

  const size_t planeX = (size_t)Bb  * IN;   // elements per X plane
  const size_t planeR = (size_t)OUT * IN;   // elements per R plane
  const size_t need   = (2 * planeX + 2 * planeR) * sizeof(unsigned short);

  if (ws_size >= need && (planeX % 2048) == 0 && (IN % 2048) == 0) {
    unsigned short* xh = (unsigned short*)d_ws;
    unsigned short* xl = xh + planeX;
    unsigned short* rh = xl + planeX;
    unsigned short* rl = rh + planeR;

    split_x_kernel<<<(unsigned)(planeX / 2048), 256, 0, stream>>>(x, xh, xl);
    split_r_kernel<<<dim3(IN / 2048, OUT), 256, 0, stream>>>(c, rh, rl, IN);
    circulant_wmma_async<<<grid, 256, 0, stream>>>(xh, xl, rh, rl, bias, out, OUT, IN);
  } else {
    circulant_wmma_fused<<<grid, 256, 0, stream>>>(x, c, bias, out, OUT, IN);
  }
}
